// Twister_14070312862266
// MI455X (gfx1250) — matrix operand order repaired and checked by reading, not compiled, not run on hardware
//
#include <hip/hip_runtime.h>
#include <hip/hip_bf16.h>
#include <math.h>

typedef __attribute__((ext_vector_type(16))) _Float16 v16h;
typedef __attribute__((ext_vector_type(8)))  float    v8f;

__device__ __forceinline__ float lrelu(float x) { return x > 0.f ? x : 0.01f * x; }

// ---------------------------------------------------------------------------
// Generic small linear: one block per row. Optional input lrelu, output LN.
// ---------------------------------------------------------------------------
template<int K, int N, bool INLRELU, bool OUTLN>
__global__ void k_linear(const float* __restrict__ X, const float* __restrict__ W,
                         const float* __restrict__ bias, const float* __restrict__ g,
                         const float* __restrict__ be, float* __restrict__ Y)
{
    __shared__ float row[K];
    __shared__ float outr[N];
    __shared__ float red[2];
    const int r = blockIdx.x;
    const float* xr = X + (size_t)r * K;
    for (int k = threadIdx.x; k < K; k += blockDim.x) {
        float v = xr[k];
        row[k] = INLRELU ? lrelu(v) : v;
    }
    __syncthreads();
    const int n = threadIdx.x;
    float acc = 0.f;
    if (n < N) {
        acc = bias[n];
        for (int k = 0; k < K; ++k) acc += row[k] * W[k * N + n];
    }
    if (OUTLN) {
        if (n < N) outr[n] = acc;
        __syncthreads();
        if (threadIdx.x == 0) {
            float mm = 0.f;
            for (int i = 0; i < N; ++i) mm += outr[i];
            mm /= (float)N;
            float vv = 0.f;
            for (int i = 0; i < N; ++i) { float d = outr[i] - mm; vv += d * d; }
            vv /= (float)N;
            red[0] = mm; red[1] = rsqrtf(vv + 1e-5f);
        }
        __syncthreads();
        if (n < N) Y[(size_t)r * N + n] = (acc - red[0]) * red[1] * g[n] + be[n];
    } else {
        if (n < N) Y[(size_t)r * N + n] = acc;
    }
}

// ---------------------------------------------------------------------------
// MPNN message: msg = concat(lrelu(h[src]), e) @ Wm + bm   (block = 128 = H)
// ---------------------------------------------------------------------------
__global__ void k_msg(const float* __restrict__ h, const float* __restrict__ e,
                      const int* __restrict__ src, const float* __restrict__ Wm,
                      const float* __restrict__ bm, float* __restrict__ msg,
                      int Nn, int E)
{
    __shared__ float row[160];
    const int be_ = blockIdx.x;
    const int b = be_ / E, eidx = be_ % E;
    const int s = src[b * E + eidx];
    const int t = threadIdx.x;
    row[t] = lrelu(h[((size_t)b * Nn + s) * 128 + t]);
    if (t < 32) row[128 + t] = e[((size_t)b * E + eidx) * 32 + t];
    __syncthreads();
    float acc = bm[t];
    for (int k = 0; k < 160; ++k) acc += row[k] * Wm[k * 128 + t];
    msg[((size_t)b * E + eidx) * 128 + t] = acc;
}

__global__ void k_scatter(const float* __restrict__ msg, const int* __restrict__ dst,
                          float* __restrict__ agg, int Nn, int E)
{
    const int be_ = blockIdx.x;
    const int b = be_ / E, eidx = be_ % E;
    const int d = dst[b * E + eidx];
    atomicAdd(&agg[((size_t)b * Nn + d) * 128 + threadIdx.x],
              msg[(size_t)be_ * 128 + threadIdx.x]);
}

// MPNN update: h' = LN(concat(lrelu(h), agg) @ Wu + bu) [+ residual]
__global__ void k_upd(const float* __restrict__ hin, const float* __restrict__ agg,
                      const float* __restrict__ Wu, const float* __restrict__ bu,
                      const float* __restrict__ g, const float* __restrict__ bl,
                      const float* __restrict__ resid, float* __restrict__ hout)
{
    __shared__ float row[256];
    __shared__ float outr[128];
    __shared__ float red[2];
    const int bn = blockIdx.x;
    const int t = threadIdx.x;
    row[t]       = lrelu(hin[(size_t)bn * 128 + t]);
    row[128 + t] = agg[(size_t)bn * 128 + t];
    __syncthreads();
    float acc = bu[t];
    for (int k = 0; k < 256; ++k) acc += row[k] * Wu[k * 128 + t];
    outr[t] = acc;
    __syncthreads();
    if (t == 0) {
        float mm = 0.f;
        for (int i = 0; i < 128; ++i) mm += outr[i];
        mm *= (1.f / 128.f);
        float vv = 0.f;
        for (int i = 0; i < 128; ++i) { float d = outr[i] - mm; vv += d * d; }
        vv *= (1.f / 128.f);
        red[0] = mm; red[1] = rsqrtf(vv + 1e-5f);
    }
    __syncthreads();
    float y = (acc - red[0]) * red[1] * g[t] + bl[t];
    if (resid) y += resid[(size_t)bn * 128 + t];
    hout[(size_t)bn * 128 + t] = y;
}

// fh_pre = LN(concat(rec_h[res_index], h2))   (block = 192)
__global__ void k_fullcat(const float* __restrict__ recH, const int* __restrict__ res_index,
                          const float* __restrict__ h2, const float* __restrict__ g,
                          const float* __restrict__ be, float* __restrict__ out)
{
    __shared__ float row[192];
    __shared__ float red[2];
    const int bi = blockIdx.x;
    const int b = bi / 320, i = bi % 320;
    const int t = threadIdx.x;
    const int ri = res_index[b * 320 + i];
    float v = (t < 128) ? recH[((size_t)b * 96 + ri) * 128 + t]
                        : h2[((size_t)b * 320 + i) * 64 + (t - 128)];
    row[t] = v;
    __syncthreads();
    if (t == 0) {
        float mm = 0.f;
        for (int k = 0; k < 192; ++k) mm += row[k];
        mm /= 192.f;
        float vv = 0.f;
        for (int k = 0; k < 192; ++k) { float d = row[k] - mm; vv += d * d; }
        vv /= 192.f;
        red[0] = mm; red[1] = rsqrtf(vv + 1e-5f);
    }
    __syncthreads();
    out[(size_t)bi * 192 + t] = (v - red[0]) * red[1] * g[t] + be[t];
}

// ---------------------------------------------------------------------------
// Pre-pack a (KSTEPS*32, 64) fp32 weight into f16 B-fragment lane layout:
// packed[((kk*4+t)*32+lane)*16 + e] = W[(kk*32 + hi*16 + e)*64 + t*16 + (lane&15)]
// (contiguous K order used here, not the ISA table: element e holds K=hi*16+e; lanes 16-31 offset K by 16)
// ---------------------------------------------------------------------------
__global__ void k_pack_b(const float* __restrict__ W, _Float16* __restrict__ out, int total)
{
    const int idx = blockIdx.x * blockDim.x + threadIdx.x;
    if (idx >= total) return;
    const int e    = idx & 15;
    const int lane = (idx >> 4) & 31;
    const int t    = (idx >> 9) & 3;
    const int kk   = idx >> 11;
    const int n = lane & 15, hi = lane >> 4;
    const int k = kk * 32 + hi * 16 + e;
    out[idx] = (_Float16)W[k * 64 + t * 16 + n];
}

// ---------------------------------------------------------------------------
// GEMM 1: inter[b,x,y,:] = lrelu(ligh[b,x,:] outer fh[b,y,:]) @ W(1024,64) + bias
// One wave per 16-row M-tile (fixed b,x; 16 consecutive y); K-block kk has
// fixed i=kk so A[m,j] = lrelu(La * fh[y,j]) with scalar La.
// A lane layout (16-bit A 16x32, contiguous as in k_pack_b): element e -> K = hi*16 + e.
// ---------------------------------------------------------------------------
__global__ void k_gemm_inter(const float* __restrict__ lighS, const float* __restrict__ fh,
                             const _Float16* __restrict__ pW, const float* __restrict__ bias,
                             float* __restrict__ inter)
{
    const int warp = (blockIdx.x * blockDim.x + threadIdx.x) >> 5;   // 0..3839
    const int y0 = (warp % 20) * 16;
    const int x  = (warp / 20) % 48;
    const int b  = warp / (20 * 48);
    const int lane = threadIdx.x & 31;
    const int m  = lane & 15;
    const int hi = (lane >> 4) & 1;
    const int y  = y0 + m;

    float fv[16];
    const float* fr = fh + ((size_t)b * 320 + y) * 32;
#pragma unroll
    for (int q = 0; q < 8; ++q) { fv[q] = fr[hi * 16 + q]; fv[8 + q] = fr[hi * 16 + 8 + q]; }

    v8f acc[4];
#pragma unroll
    for (int t = 0; t < 4; ++t)
#pragma unroll
        for (int q = 0; q < 8; ++q) acc[t][q] = 0.f;

    const float* lr = lighS + ((size_t)b * 48 + x) * 32;
    const v16h* pWv = (const v16h*)pW;
    for (int kk = 0; kk < 32; ++kk) {
        const float La = lr[kk];
        v16h a;
#pragma unroll
        for (int e = 0; e < 16; ++e) a[e] = (_Float16)lrelu(La * fv[e]);
#pragma unroll
        for (int t = 0; t < 4; ++t) {
            v16h bb = pWv[(kk * 4 + t) * 32 + lane];
            acc[t] = __builtin_amdgcn_wmma_f32_16x16x32_f16(false, a, false, bb,
                                                            (short)0, acc[t], false, false);
        }
    }
    // D layout: row M = hi*8 + r, col N = lane&15
    const size_t rowbase = (size_t)(b * 48 + x) * 320;
#pragma unroll
    for (int t = 0; t < 4; ++t) {
        const int o = t * 16 + m;
        const float bo = bias[o];
#pragma unroll
        for (int rr = 0; rr < 8; ++rr) {
            const int yy = y0 + hi * 8 + rr;
            inter[(rowbase + yy) * 64 + o] = acc[t][rr] + bo;
        }
    }
}

// flat attention logits: one thread per row (bias dropped: softmax shift-invariant)
__global__ void k_flat_logits(const float* __restrict__ inter, const float* __restrict__ atnW,
                              float* __restrict__ logits)
{
    const int row = blockIdx.x * blockDim.x + threadIdx.x;
    if (row >= 61440) return;
    const float* r = inter + (size_t)row * 64;
    float s = 0.f;
    for (int k = 0; k < 64; ++k) s += r[k] * atnW[k];
    logits[row] = s;
}

// in_mat = lrelu(inter) @ din_W(64,8) + din_b
__global__ void k_inmat(const float* __restrict__ inter, const float* __restrict__ W,
                        const float* __restrict__ b8, float* __restrict__ out)
{
    const int row = blockIdx.x * blockDim.x + threadIdx.x;
    if (row >= 61440) return;
    const float* r = inter + (size_t)row * 64;
    float acc[8];
#pragma unroll
    for (int i = 0; i < 8; ++i) acc[i] = b8[i];
    for (int k = 0; k < 64; ++k) {
        const float v = lrelu(r[k]);
#pragma unroll
        for (int i = 0; i < 8; ++i) acc[i] += v * W[k * 8 + i];
    }
#pragma unroll
    for (int i = 0; i < 8; ++i) out[(size_t)row * 8 + i] = acc[i];
}

// softmax stats (max, sumexp) per group
__global__ void k_stats(const float* __restrict__ logits, float* __restrict__ stats, int gsz)
{
    __shared__ float sm[256];
    const int g = blockIdx.x;
    const float* l = logits + (size_t)g * gsz;
    const int t = threadIdx.x;
    float mx = -1e30f;
    for (int i = t; i < gsz; i += 256) mx = fmaxf(mx, l[i]);
    sm[t] = mx; __syncthreads();
    for (int s = 128; s > 0; s >>= 1) { if (t < s) sm[t] = fmaxf(sm[t], sm[t + s]); __syncthreads(); }
    const float M = sm[0];
    __syncthreads();
    float se = 0.f;
    for (int i = t; i < gsz; i += 256) se += expf(l[i] - M);
    sm[t] = se; __syncthreads();
    for (int s = 128; s > 0; s >>= 1) { if (t < s) sm[t] += sm[t + s]; __syncthreads(); }
    if (t == 0) { stats[g * 2] = M; stats[g * 2 + 1] = sm[0]; }
}

// flat pooled_pre[b,:] = sum_rows softmax(logit)_row * inter[row,:]
__global__ void k_flat_pool(const float* __restrict__ inter, const float* __restrict__ logits,
                            const float* __restrict__ stats, float* __restrict__ pooled)
{
    __shared__ float sm[256];
    const int b = blockIdx.x;
    const int t = threadIdx.x;
    const int ch = t & 63, grp = t >> 6;
    const float M = stats[b * 2];
    const float inv = 1.f / stats[b * 2 + 1];
    float acc = 0.f;
    for (int row = grp; row < 15360; row += 4) {
        const float wv = expf(logits[b * 15360 + row] - M) * inv;
        acc += wv * inter[((size_t)b * 15360 + row) * 64 + ch];
    }
    sm[t] = acc; __syncthreads();
    if (t < 64) pooled[b * 64 + t] = sm[t] + sm[t + 64] + sm[t + 128] + sm[t + 192];
}

// head: pooled = pooled_pre @ out_W + out_b; LN(lrelu(pooled) @ head_W + head_b)
__global__ void k_head(const float* __restrict__ pooled, const float* __restrict__ outW,
                       const float* __restrict__ outb, const float* __restrict__ headW,
                       const float* __restrict__ headb, const float* __restrict__ lng,
                       const float* __restrict__ lnb, float* __restrict__ full, int isDist)
{
    __shared__ float pv[64];
    __shared__ float h3[3];
    const int g = blockIdx.x;
    const int t = threadIdx.x;
    const float* pp = pooled + g * 64;
    float acc = outb[t];
    for (int k = 0; k < 64; ++k) acc += pp[k] * outW[k * 64 + t];
    pv[t] = lrelu(acc);
    __syncthreads();
    if (t < 3) {
        float s = headb[t];
        for (int k = 0; k < 64; ++k) s += pv[k] * headW[k * 3 + t];
        h3[t] = s;
    }
    __syncthreads();
    if (t == 0) {
        const float mm = (h3[0] + h3[1] + h3[2]) / 3.f;
        const float vv = ((h3[0]-mm)*(h3[0]-mm) + (h3[1]-mm)*(h3[1]-mm) + (h3[2]-mm)*(h3[2]-mm)) / 3.f;
        const float rs = rsqrtf(vv + 1e-5f);
        int bb, slot;
        if (isDist) { bb = g >> 2; slot = 1 + (g & 3); } else { bb = g; slot = 0; }
        for (int c = 0; c < 3; ++c)
            full[(bb * 5 + slot) * 3 + c] = (h3[c] - mm) * rs * lng[c] + lnb[c];
    }
}

// ---------------------------------------------------------------------------
// GEMM 2 (two passes, dmat never materialized):
// dmat[row,:] = lrelu(in_mat[row,:8] outer rbf[row,:16]) @ dout_W(128,64) + dout_b
// PASS 1: attention logits per row (16-lane shfl-xor reduce) -> logits
// PASS 2: recompute tile, accumulate softmax-weighted row sum into pooled (LDS+atomics)
// ---------------------------------------------------------------------------
template<int PASS>
__global__ void k_gemm_dist(const float* __restrict__ in_mat, const float* __restrict__ pose_coord,
                            const float* __restrict__ rec_coord, const _Float16* __restrict__ pW,
                            const float* __restrict__ dout_b, const float* __restrict__ atnW,
                            float* __restrict__ logits, const float* __restrict__ stats,
                            float* __restrict__ pooled)
{
    __shared__ float sm[64];
    const int bp = blockIdx.y;                       // b*4 + p
    const int b = bp >> 2;
    const int tile = blockIdx.x * 4 + (threadIdx.x >> 5);  // 0..959
    const int x  = tile / 20;
    const int r0 = (tile % 20) * 16;
    const int lane = threadIdx.x & 31;
    const int m  = lane & 15;
    const int hi = (lane >> 4) & 1;
    const int r  = r0 + m;

    if (PASS == 2) {
        if (threadIdx.x < 64) sm[threadIdx.x] = 0.f;
        __syncthreads();
    }

    const float* pc = pose_coord + ((size_t)bp * 48 + x) * 3;
    const float* rc = rec_coord + ((size_t)b * 320 + r) * 3;
    const float dx = pc[0] - rc[0], dy = pc[1] - rc[1], dz = pc[2] - rc[2];
    const float dist = sqrtf(dx * dx + dy * dy + dz * dz + 1e-10f);
    float rbf[16];
#pragma unroll
    for (int j = 0; j < 16; ++j) {
        const float u = (dist - (float)j * (8.0f / 15.0f)) * 2.0f;   // /spacing(0.5)
        rbf[j] = expf(-u * u);
    }
    float im[8];
    const float* imr = in_mat + ((size_t)(b * 48 + x) * 320 + r) * 8;
#pragma unroll
    for (int i = 0; i < 8; ++i) im[i] = imr[i];

    v8f acc[4];
#pragma unroll
    for (int t = 0; t < 4; ++t)
#pragma unroll
        for (int q = 0; q < 8; ++q) acc[t][q] = 0.f;

    const v16h* pWv = (const v16h*)pW;
#pragma unroll
    for (int kk = 0; kk < 4; ++kk) {
        v16h a;
#pragma unroll
        for (int e = 0; e < 16; ++e) {
            const int q = hi * 16 + e;
            const int k = kk * 32 + q;                 // K index = i*16 + j
            a[e] = (_Float16)lrelu(im[k >> 4] * rbf[k & 15]);
        }
#pragma unroll
        for (int t = 0; t < 4; ++t) {
            v16h bb = pWv[(kk * 4 + t) * 32 + lane];
            acc[t] = __builtin_amdgcn_wmma_f32_16x16x32_f16(false, a, false, bb,
                                                            (short)0, acc[t], false, false);
        }
    }

    if (PASS == 1) {
        float part[8];
#pragma unroll
        for (int rr = 0; rr < 8; ++rr) {
            float s = 0.f;
#pragma unroll
            for (int t = 0; t < 4; ++t) s += (acc[t][rr] + dout_b[t * 16 + m]) * atnW[t * 16 + m];
            part[rr] = s;
        }
#pragma unroll
        for (int offm = 1; offm < 16; offm <<= 1) {
#pragma unroll
            for (int rr = 0; rr < 8; ++rr) part[rr] += __shfl_xor(part[rr], offm, 32);
        }
        if (m == 0) {
#pragma unroll
            for (int rr = 0; rr < 8; ++rr) {
                const int row = x * 320 + r0 + hi * 8 + rr;
                logits[(size_t)bp * 15360 + row] = part[rr];
            }
        }
    } else {
        const float M = stats[bp * 2];
        const float inv = 1.f / stats[bp * 2 + 1];
        float wgt[8];
#pragma unroll
        for (int rr = 0; rr < 8; ++rr) {
            const int row = x * 320 + r0 + hi * 8 + rr;
            wgt[rr] = expf(logits[(size_t)bp * 15360 + row] - M) * inv;
        }
#pragma unroll
        for (int t = 0; t < 4; ++t) {
            float c = 0.f;
#pragma unroll
            for (int rr = 0; rr < 8; ++rr) c += wgt[rr] * (acc[t][rr] + dout_b[t * 16 + m]);
            atomicAdd(&sm[t * 16 + m], c);
        }
        __syncthreads();
        if (threadIdx.x < 64) atomicAdd(&pooled[bp * 64 + threadIdx.x], sm[threadIdx.x]);
    }
}

// final: softmax over attention column, assemble all 6 outputs (84 floats)
__global__ void k_final(const float* __restrict__ full, float* __restrict__ out)
{
    const int b = threadIdx.x;
    if (b >= 4) return;
    float preds[5], atn[5], actp[5];
    for (int q = 0; q < 5; ++q) {
        preds[q] = full[(b * 5 + q) * 3 + 0];
        atn[q]   = full[(b * 5 + q) * 3 + 1];
        actp[q]  = full[(b * 5 + q) * 3 + 2];
    }
    float M = atn[0];
    for (int q = 1; q < 5; ++q) M = fmaxf(M, atn[q]);
    float s = 0.f, a[5];
    for (int q = 0; q < 5; ++q) { a[q] = expf(atn[q] - M); s += a[q]; }
    float score = 0.f, act = 0.f;
    for (int q = 0; q < 5; ++q) { a[q] /= s; score += preds[q] * a[q]; act += actp[q] * a[q]; }
    out[b] = score;
    out[4 + b] = act;
    for (int p = 0; p < 4; ++p) out[8 + b * 4 + p] = full[(b * 5 + 1 + p) * 3 + 1];
    for (int q = 0; q < 5; ++q) out[24 + b * 5 + q] = atn[q];
    for (int q = 0; q < 5; ++q) out[44 + b * 5 + q] = preds[q];
    for (int q = 0; q < 5; ++q) out[64 + b * 5 + q] = actp[q];
}

// ---------------------------------------------------------------------------
extern "C" void kernel_launch(void* const* d_in, const int* in_sizes, int n_in,
                              void* d_out, int out_size, void* d_ws, size_t ws_size,
                              hipStream_t stream)
{
    const float* lig_x      = (const float*)d_in[0];
    const float* lig_ef     = (const float*)d_in[1];
    const int*   lig_src    = (const int*)d_in[2];
    const int*   lig_dst    = (const int*)d_in[3];
    const float* rec_x      = (const float*)d_in[4];
    const float* rec_ef     = (const float*)d_in[5];
    const int*   rec_src    = (const int*)d_in[6];
    const int*   rec_dst    = (const int*)d_in[7];
    const float* full_x     = (const float*)d_in[8];
    const int*   res_index  = (const int*)d_in[9];
    const float* rec_coord  = (const float*)d_in[10];
    const float* pose_coord = (const float*)d_in[11];

    // parameter sizes in _make_params insertion order
    static const int psz[84] = {
        2048,128,256,32,16384,128,128,128,
        20480,128,32768,128,128,128,
        20480,128,32768,128,128,128,
        20480,128,32768,128,128,128,          // lig (0..25)
        2048,128,256,32,16384,128,128,128,
        20480,128,32768,128,128,128,
        20480,128,32768,128,128,128,
        20480,128,32768,128,128,128,          // rec (26..51)
        4096,32,32,32,                        // lig_out (52..55)
        1024,64,                              // full_emb (56,57)
        192,192,                              // full_ln (58,59)
        6144,32,                              // full_out (60,61)
        65536,64,                             // inter (62,63)
        512,8,                                // din (64,65)
        8192,64,                              // dout (66,67)
        64,1,4096,64,192,3,3,3,               // flat head (68..75)
        64,1,4096,64,192,3,3,3                // dist head (76..83)
    };
    const float* P[84];
    if (n_in >= 12 + 84) {
        for (int i = 0; i < 84; ++i) P[i] = (const float*)d_in[12 + i];
    } else {
        const float* base = (const float*)d_in[12];
        size_t off = 0;
        for (int i = 0; i < 84; ++i) { P[i] = base + off; off += psz[i]; }
    }

    // workspace carve (all offsets multiples of 8 floats -> 32B aligned)
    float* w = (float*)d_ws;
    size_t off = 0;
    auto A = [&](size_t n) -> float* { float* p = w + off; off += n; return p; };
    float* ebuf    = A((size_t)4*768*32);
    float* hA      = A((size_t)4*96*128);
    float* hB      = A((size_t)4*96*128);
    float* hC      = A((size_t)4*96*128);
    float* msgb    = A((size_t)4*768*128);
    float* aggb    = A((size_t)4*96*128);
    float* lighS   = A((size_t)4*48*32);
    float* fullh2  = A((size_t)4*320*64);
    float* fhpre   = A((size_t)4*320*192);
    float* fh      = A((size_t)4*320*32);
    _Float16* packInter = (_Float16*)A(65536/2);
    _Float16* packDout  = (_Float16*)A(8192/2);
    float* inter   = A((size_t)4*48*320*64);
    float* in_mat  = A((size_t)4*48*320*8);
    float* logitF  = A((size_t)4*15360);
    float* logitD  = A((size_t)16*15360);
    float* statsF  = A(8);
    float* statsD  = A(32);
    float* pooledF = A(4*64);
    float* pooledD = A(16*64);
    float* fullb   = A(64);

    auto encode = [&](const float* x, const float* ef, const int* src, const int* dst,
                      int Nn, int E, const float* const* Pp) -> const float* {
        k_linear<16,128,false,false><<<4*Nn,128,0,stream>>>(x, Pp[0], Pp[1], nullptr, nullptr, hA);
        k_linear<8,32,false,false><<<4*E,64,0,stream>>>(ef, Pp[2], Pp[3], nullptr, nullptr, ebuf);
        k_linear<128,128,true,true><<<4*Nn,128,0,stream>>>(hA, Pp[4], Pp[5], Pp[6], Pp[7], hA);
        const float* cur = hA;
        float* outs[3] = { hB, hC, hA };
        const float* resid[3] = { nullptr, nullptr, hB };
        for (int l = 0; l < 3; ++l) {
            k_msg<<<4*E,128,0,stream>>>(cur, ebuf, src, Pp[8+l*6], Pp[9+l*6], msgb, Nn, E);
            hipMemsetAsync(aggb, 0, (size_t)4*Nn*128*sizeof(float), stream);
            k_scatter<<<4*E,128,0,stream>>>(msgb, dst, aggb, Nn, E);
            k_upd<<<4*Nn,128,0,stream>>>(cur, aggb, Pp[10+l*6], Pp[11+l*6],
                                         Pp[12+l*6], Pp[13+l*6], resid[l], outs[l]);
            cur = outs[l];
        }
        return cur;
    };

    // ligand encoder + out projection (lig buffers free after lighS)
    const float* ligH = encode(lig_x, lig_ef, lig_src, lig_dst, 48, 128, &P[0]);
    k_linear<128,32,true,true><<<4*48,128,0,stream>>>(ligH, P[52], P[53], P[54], P[55], lighS);

    // receptor encoder (reuses the same buffers)
    const float* recH = encode(rec_x, rec_ef, rec_src, rec_dst, 96, 768, &P[26]);

    // full-receptor features
    k_linear<16,64,false,false><<<4*320,64,0,stream>>>(full_x, P[56], P[57], nullptr, nullptr, fullh2);
    k_fullcat<<<4*320,192,0,stream>>>(recH, res_index, fullh2, P[58], P[59], fhpre);
    k_linear<192,32,true,false><<<4*320,192,0,stream>>>(fhpre, P[60], P[61], nullptr, nullptr, fh);

    // pack weights into WMMA B-fragment layout (f16, L2 resident)
    k_pack_b<<<65536/256,256,0,stream>>>(P[62], packInter, 65536);
    k_pack_b<<<8192/256,256,0,stream>>>(P[66], packDout, 8192);

    // big GEMM 1: inter (8 GFLOP, WMMA)
    k_gemm_inter<<<960,128,0,stream>>>(lighS, fh, packInter, P[63], inter);

    // flat pool head (factored) + in_mat
    k_flat_logits<<<(61440+255)/256,256,0,stream>>>(inter, P[68], logitF);
    k_inmat<<<(61440+255)/256,256,0,stream>>>(inter, P[64], P[65], in_mat);
    k_stats<<<4,256,0,stream>>>(logitF, statsF, 15360);
    k_flat_pool<<<4,256,0,stream>>>(inter, logitF, statsF, pooledF);
    k_head<<<4,64,0,stream>>>(pooledF, P[70], P[71], P[72], P[73], P[74], P[75], fullb, 0);

    // big GEMM 2: dist path, two WMMA passes (logits, then weighted pooling)
    k_gemm_dist<1><<<dim3(240,16),128,0,stream>>>(in_mat, pose_coord, rec_coord, packDout,
                                                  P[67], P[76], logitD, nullptr, nullptr);
    k_stats<<<16,256,0,stream>>>(logitD, statsD, 15360);
    hipMemsetAsync(pooledD, 0, 16*64*sizeof(float), stream);
    k_gemm_dist<2><<<dim3(240,16),128,0,stream>>>(in_mat, pose_coord, rec_coord, packDout,
                                                  P[67], P[76], logitD, statsD, pooledD);
    k_head<<<16,64,0,stream>>>(pooledD, P[78], P[79], P[80], P[81], P[82], P[83], fullb, 1);

    k_final<<<1,32,0,stream>>>(fullb, (float*)d_out);

    (void)in_sizes; (void)out_size; (void)ws_size;
}
